// CrossAttentionBlock_15650860827496
// MI455X (gfx1250) — matrix operand order repaired and checked by reading, not compiled, not run on hardware
//
#include <hip/hip_runtime.h>
#include <hip/hip_bf16.h>
#include <math.h>
#include <stdint.h>

// Problem constants (match reference)
#define BB 2
#define NN 2048
#define CC 1024
#define HH 16
#define HDIM 64
#define FFD (2 * CC)
#define MTOT (BB * NN) // 4096 rows

typedef __attribute__((ext_vector_type(16))) __bf16 v16bf;
typedef __attribute__((ext_vector_type(8))) float v8f;

union Frag {
    v16bf h;
    unsigned int u[8];
};

union DwordPair {
    unsigned int u;
    __bf16 h[2];
};

__device__ __forceinline__ v8f wmma_bf16(const Frag& a, const Frag& b, v8f c) {
    return __builtin_amdgcn_wmma_f32_16x16x32_bf16(false, a.h, false, b.h,
                                                   (short)0, c, false, false);
}

// A-operand (16x32 bf16) K index for VGPR j, lane-half: contiguous k = 16*half + 2*j
__device__ __forceinline__ int a_kidx(int j, int half) {
    return 2 * j + 16 * half;
}

// CDNA5 async copy: 16B global -> LDS, tracked by ASYNCcnt (no VGPR round trip).
__device__ __forceinline__ void async_ld_b128(unsigned lds_off, const void* gptr) {
    unsigned long long ga = (unsigned long long)(uintptr_t)gptr;
    asm volatile("global_load_async_to_lds_b128 %0, %1, off"
                 :: "v"(lds_off), "v"(ga) : "memory");
}
__device__ __forceinline__ void wait_async0() {
    asm volatile("s_wait_asynccnt 0" ::: "memory");
}
__device__ __forceinline__ unsigned lds_off(const void* p) {
    return (unsigned)(uintptr_t)p; // low 32 bits of LDS aperture address = LDS offset
}

// ---------------------------------------------------------------- f32 -> bf16
__global__ void cvt_bf16(const float* __restrict__ in, __bf16* __restrict__ out, int n) {
    int i = blockIdx.x * blockDim.x + threadIdx.x;
    if (i < n) out[i] = (__bf16)in[i];
}

// ------------------------------------------------------------------- GEMM
// C[M,N] = act(A[M,K](bf16) x B[K,N](bf16) + bias) (+ res).
// Block tile 128x256, 8 waves (2x4), wave tile 64x64, K-step 32.
// A staged via async-to-LDS; B staged transposed [n][k] for contiguous b128 frags.
template <int OUT_BF16, int ACT, int RES>
__global__ void __launch_bounds__(256) gemm_wmma(
    const __bf16* __restrict__ A, const __bf16* __restrict__ Bw,
    const float* __restrict__ bias, const float* __restrict__ res,
    float* __restrict__ outF, __bf16* __restrict__ outH,
    int M, int N, int K) {
    constexpr int LDA = 40; // padded row stride (elems): 80B, 16B-aligned, bank-clean
    constexpr int LDB = 40;
    alignas(16) __shared__ __bf16 sA[128 * LDA];  // [m][k]
    alignas(16) __shared__ __bf16 sBt[256 * LDB]; // [n][k] (transposed)

    const int lane = threadIdx.x & 31;
    const int wave = threadIdx.x >> 5;
    const int half = lane >> 4, l16 = lane & 15;
    const int wm = wave >> 2, wn = wave & 3; // 2 x 4 wave grid
    const int tilesN = N >> 8;
    const int bm = blockIdx.x / tilesN, bn = blockIdx.x % tilesN;
    const int m0b = bm << 7, n0b = bn << 8;

    const unsigned sAo = lds_off(sA);

    v8f acc[4][4];
#pragma unroll
    for (int i = 0; i < 4; ++i)
#pragma unroll
        for (int j = 0; j < 4; ++j) acc[i][j] = (v8f)0.0f;

    for (int k0 = 0; k0 < K; k0 += 32) {
        // --- stage A tile 128x32 (8KB = 512 x 16B chunks; 2 per thread), async
#pragma unroll
        for (int i = 0; i < 2; ++i) {
            const int chunk = threadIdx.x + i * 256;
            const int r = chunk >> 2, c4 = chunk & 3;
            async_ld_b128(sAo + (unsigned)(r * (LDA * 2) + c4 * 16),
                          A + (size_t)(m0b + r) * K + k0 + c4 * 8);
        }
        // --- stage B tile 32x256 transposed: sBt[n][k] = Bw[k0+k][n0b+n]
        // coalesced dword reads along n, 16 per thread
#pragma unroll
        for (int i = 0; i < 16; ++i) {
            const int idx = threadIdx.x + i * 256;
            const int kk = idx >> 7, c2 = idx & 127;
            DwordPair d;
            d.u = *reinterpret_cast<const unsigned int*>(
                Bw + (size_t)(k0 + kk) * N + n0b + 2 * c2);
            sBt[(2 * c2) * LDB + kk] = d.h[0];
            sBt[(2 * c2 + 1) * LDB + kk] = d.h[1];
        }
        wait_async0();
        __syncthreads();

        // --- fragments from LDS (contiguous b128 groups)
        Frag a[4], b[4];
#pragma unroll
        for (int i = 0; i < 4; ++i) {
            const __bf16* ap = &sA[(wm * 64 + 16 * i + l16) * LDA];
#pragma unroll
            for (int j = 0; j < 8; ++j)
                a[i].u[j] = *reinterpret_cast<const unsigned int*>(ap + a_kidx(j, half));
        }
#pragma unroll
        for (int i = 0; i < 4; ++i) {
            const __bf16* bp = &sBt[(wn * 64 + 16 * i + l16) * LDB + 16 * half];
#pragma unroll
            for (int j = 0; j < 8; ++j)
                b[i].u[j] = *reinterpret_cast<const unsigned int*>(bp + 2 * j);
        }
#pragma unroll
        for (int i = 0; i < 4; ++i)
#pragma unroll
            for (int j = 0; j < 4; ++j) acc[i][j] = wmma_bf16(a[i], b[j], acc[i][j]);
        __syncthreads();
    }

    const int m0 = m0b + wm * 64, n0 = n0b + wn * 64;
#pragma unroll
    for (int i = 0; i < 4; ++i)
#pragma unroll
        for (int j = 0; j < 4; ++j)
#pragma unroll
            for (int r = 0; r < 8; ++r) {
                const int m = m0 + 16 * i + r + 8 * half;
                const int n = n0 + 16 * j + l16;
                float v = acc[i][j][r] + bias[n];
                if (ACT == 1) v = 0.5f * v * (1.0f + erff(v * 0.70710678118654752f));
                if (RES) v += res[(size_t)m * N + n];
                if (OUT_BF16)
                    outH[(size_t)m * N + n] = (__bf16)v;
                else
                    outF[(size_t)m * N + n] = v;
            }
}

// --------------------------------------------------------------- Attention
// 4 waves / block share one (b,h); each wave owns 16 queries, flash-style.
__global__ void __launch_bounds__(128) attn_wmma(
    const __bf16* __restrict__ qh, const __bf16* __restrict__ kh,
    const __bf16* __restrict__ vh, __bf16* __restrict__ ctx) {
    constexpr int LKV = 72; // padded row stride (144B, 16B-aligned, bank-clean)
    alignas(16) __shared__ __bf16 sK[64 * LKV];    // [key][c]
    alignas(16) __shared__ __bf16 sV[64 * LKV];    // [key][c]
    alignas(16) __shared__ __bf16 sP[4][16 * LKV]; // per-wave P staging [row][key]

    const int lane = threadIdx.x & 31;
    const int wave = threadIdx.x >> 5;
    const int half = lane >> 4, l16 = lane & 15;
    const int qt = blockIdx.x % (NN / 64);
    const int bh = blockIdx.x / (NN / 64);
    const int b = bh / HH, h = bh % HH;
    const int q0 = qt * 64 + wave * 16;
    const float scale = 0.125f; // HDIM^-0.5

    const unsigned sKo = lds_off(sK);
    const unsigned sVo = lds_off(sV);

    // Q as two A-fragments (c = 0..31, 32..63), loaded once from global (L2-hot)
    Frag aq[2];
    {
        const __bf16* qp = qh + (size_t)(b * NN + q0 + l16) * CC + h * HDIM;
#pragma unroll
        for (int ks = 0; ks < 2; ++ks)
#pragma unroll
            for (int j = 0; j < 8; ++j)
                aq[ks].u[j] =
                    *reinterpret_cast<const unsigned int*>(qp + ks * 32 + a_kidx(j, half));
    }

    v8f o[4];
#pragma unroll
    for (int f = 0; f < 4; ++f) o[f] = (v8f)0.0f;
    float mrow[8], lrow[8];
#pragma unroll
    for (int r = 0; r < 8; ++r) { mrow[r] = -1e30f; lrow[r] = 0.0f; }

    for (int kt = 0; kt < NN / 64; ++kt) {
        const int key0 = kt * 64;
        // Async-stage K/V tiles (128 keys x 32 ch bf16 = 512 x 16B chunks each)
        {
            const size_t baseKV = (size_t)(b * NN + key0) * CC + h * HDIM;
#pragma unroll
            for (int i = 0; i < 4; ++i) {
                const int chunk = threadIdx.x + i * 128;
                const int r = chunk >> 2, c4 = chunk & 3;
                const size_t goff = baseKV + (size_t)r * CC + c4 * 8;
                const unsigned loff = (unsigned)(r * (LKV * 2) + c4 * 16);
                async_ld_b128(sKo + loff, kh + goff);
                async_ld_b128(sVo + loff, vh + goff);
            }
        }
        wait_async0();
        __syncthreads();

        // S = Q x K^T : 4 fragments over 64 keys, 2 K-steps over 64 channels
        v8f s[4];
#pragma unroll
        for (int f = 0; f < 4; ++f) s[f] = (v8f)0.0f;
#pragma unroll
        for (int ks = 0; ks < 2; ++ks) {
#pragma unroll
            for (int f = 0; f < 4; ++f) {
                Frag bk;
#pragma unroll
                for (int j = 0; j < 8; ++j) {
                    const int c = ks * 32 + 16 * half + 2 * j;
                    bk.u[j] = *reinterpret_cast<const unsigned int*>(
                        &sK[(16 * f + l16) * LKV + c]);
                }
                s[f] = wmma_bf16(aq[ks], bk, s[f]);
            }
        }

        // Online softmax per D-fragment row (row m = r + 8*half)
#pragma unroll
        for (int r = 0; r < 8; ++r) {
            float tmax = -1e30f;
#pragma unroll
            for (int f = 0; f < 4; ++f) {
                s[f][r] *= scale;
                tmax = fmaxf(tmax, s[f][r]);
            }
#pragma unroll
            for (int msk = 1; msk < 16; msk <<= 1)
                tmax = fmaxf(tmax, __shfl_xor(tmax, msk, 32));
            const float mn = fmaxf(mrow[r], tmax);
            const float alpha = __expf(mrow[r] - mn);
            float rsum = 0.0f;
#pragma unroll
            for (int f = 0; f < 4; ++f) {
                const float p = __expf(s[f][r] - mn);
                rsum += p;
                sP[wave][(r + 8 * half) * LKV + 16 * f + l16] = (__bf16)p;
            }
#pragma unroll
            for (int msk = 1; msk < 16; msk <<= 1) rsum += __shfl_xor(rsum, msk, 32);
            lrow[r] = lrow[r] * alpha + rsum;
            mrow[r] = mn;
#pragma unroll
            for (int f = 0; f < 4; ++f) o[f][r] *= alpha;
        }
        // wave-local DS store->load sync for sP
        asm volatile("s_wait_dscnt 0" ::: "memory");

        // O += P x V : contraction over 64 keys, 4 output channel fragments
#pragma unroll
        for (int ks = 0; ks < 2; ++ks) {
            Frag ap;
#pragma unroll
            for (int j = 0; j < 8; ++j)
                ap.u[j] = *reinterpret_cast<const unsigned int*>(
                    &sP[wave][l16 * LKV + ks * 32 + a_kidx(j, half)]);
#pragma unroll
            for (int f = 0; f < 4; ++f) {
                Frag bv;
#pragma unroll
                for (int j = 0; j < 8; ++j) {
                    const int key = ks * 32 + 16 * half + 2 * j;
                    bv.h[2 * j] = sV[key * LKV + 16 * f + l16];
                    bv.h[2 * j + 1] = sV[(key + 1) * LKV + 16 * f + l16];
                }
                o[f] = wmma_bf16(ap, bv, o[f]);
            }
        }
        __syncthreads();
    }

    // Normalize and write ctx (bf16, [B,N,C] with head offset)
#pragma unroll
    for (int f = 0; f < 4; ++f)
#pragma unroll
        for (int r = 0; r < 8; ++r) {
            const int m = q0 + r + 8 * half;
            const int c = h * HDIM + 16 * f + l16;
            ctx[(size_t)(b * NN + m) * CC + c] = (__bf16)(o[f][r] / lrow[r]);
        }
}

// --------------------------------------------------------------- LayerNorm
__global__ void __launch_bounds__(256) ln_kernel(
    const float* __restrict__ x, const float* __restrict__ w,
    const float* __restrict__ bsrc, __bf16* __restrict__ out) {
    __shared__ float red[256];
    const int row = blockIdx.x;
    const float* xr = x + (size_t)row * CC;
    float s = 0.0f, s2 = 0.0f;
    for (int i = threadIdx.x; i < CC; i += 256) {
        const float v = xr[i];
        s += v;
        s2 += v * v;
    }
    red[threadIdx.x] = s;
    __syncthreads();
    for (int st = 128; st > 0; st >>= 1) {
        if (threadIdx.x < st) red[threadIdx.x] += red[threadIdx.x + st];
        __syncthreads();
    }
    const float mu = red[0] * (1.0f / CC);
    __syncthreads();
    red[threadIdx.x] = s2;
    __syncthreads();
    for (int st = 128; st > 0; st >>= 1) {
        if (threadIdx.x < st) red[threadIdx.x] += red[threadIdx.x + st];
        __syncthreads();
    }
    const float var = red[0] * (1.0f / CC) - mu * mu;
    const float inv = rsqrtf(var + 1e-5f);
    for (int i = threadIdx.x; i < CC; i += 256)
        out[(size_t)row * CC + i] = (__bf16)((xr[i] - mu) * inv * w[i] + bsrc[i]);
}

// ------------------------------------------------------------------ driver
extern "C" void kernel_launch(void* const* d_in, const int* in_sizes, int n_in,
                              void* d_out, int out_size, void* d_ws, size_t ws_size,
                              hipStream_t stream) {
    (void)in_sizes; (void)n_in; (void)out_size; (void)ws_size;
    const float* q   = (const float*)d_in[0];
    const float* k   = (const float*)d_in[1];
    const float* v   = (const float*)d_in[2];
    const float* Wq  = (const float*)d_in[3];
    const float* bq  = (const float*)d_in[4];
    const float* Wk  = (const float*)d_in[5];
    const float* bk  = (const float*)d_in[6];
    const float* Wv  = (const float*)d_in[7];
    const float* bv  = (const float*)d_in[8];
    const float* Wo  = (const float*)d_in[9];
    const float* bo  = (const float*)d_in[10];
    const float* lnw = (const float*)d_in[11];
    const float* lnb = (const float*)d_in[12];
    const float* W1  = (const float*)d_in[13];
    const float* b1  = (const float*)d_in[14];
    const float* W2  = (const float*)d_in[15];
    const float* b2  = (const float*)d_in[16];
    float* out = (float*)d_out;

    char* ws = (char*)d_ws;
    size_t off = 0;
    auto alloc = [&](size_t bytes) -> void* {
        void* p = ws + off;
        off += (bytes + 255) & ~(size_t)255;
        return p;
    };
    const size_t ACT_E = (size_t)MTOT * CC; // 4M elements
    __bf16* qb   = (__bf16*)alloc(ACT_E * 2);
    __bf16* kb   = (__bf16*)alloc(ACT_E * 2);
    __bf16* vb   = (__bf16*)alloc(ACT_E * 2);
    __bf16* Wqb  = (__bf16*)alloc((size_t)CC * CC * 2);
    __bf16* Wkb  = (__bf16*)alloc((size_t)CC * CC * 2);
    __bf16* Wvb  = (__bf16*)alloc((size_t)CC * CC * 2);
    __bf16* Wob  = (__bf16*)alloc((size_t)CC * CC * 2);
    __bf16* W1b  = (__bf16*)alloc((size_t)CC * FFD * 2);
    __bf16* W2b  = (__bf16*)alloc((size_t)FFD * CC * 2);
    __bf16* qhB  = (__bf16*)alloc(ACT_E * 2);
    __bf16* khB  = (__bf16*)alloc(ACT_E * 2);
    __bf16* vhB  = (__bf16*)alloc(ACT_E * 2);
    __bf16* ctxB = (__bf16*)alloc(ACT_E * 2);
    float*  xF   = (float*)alloc(ACT_E * 4);
    __bf16* hB   = (__bf16*)alloc(ACT_E * 2);
    __bf16* gB   = (__bf16*)alloc((size_t)MTOT * FFD * 2);

    // bf16 conversions
    const int nAct = MTOT * CC;
    cvt_bf16<<<(nAct + 255) / 256, 256, 0, stream>>>(q, qb, nAct);
    cvt_bf16<<<(nAct + 255) / 256, 256, 0, stream>>>(k, kb, nAct);
    cvt_bf16<<<(nAct + 255) / 256, 256, 0, stream>>>(v, vb, nAct);
    const int nW = CC * CC;
    cvt_bf16<<<(nW + 255) / 256, 256, 0, stream>>>(Wq, Wqb, nW);
    cvt_bf16<<<(nW + 255) / 256, 256, 0, stream>>>(Wk, Wkb, nW);
    cvt_bf16<<<(nW + 255) / 256, 256, 0, stream>>>(Wv, Wvb, nW);
    cvt_bf16<<<(nW + 255) / 256, 256, 0, stream>>>(Wo, Wob, nW);
    const int nW1 = CC * FFD;
    cvt_bf16<<<(nW1 + 255) / 256, 256, 0, stream>>>(W1, W1b, nW1);
    cvt_bf16<<<(nW1 + 255) / 256, 256, 0, stream>>>(W2, W2b, nW1);

    // QKV projections (bf16 out). Block tile 128x256.
    const int blkCC = (MTOT / 128) * (CC / 256); // 128
    gemm_wmma<1, 0, 0><<<blkCC, 256, 0, stream>>>(qb, Wqb, bq, nullptr, nullptr, qhB, MTOT, CC, CC);
    gemm_wmma<1, 0, 0><<<blkCC, 256, 0, stream>>>(kb, Wkb, bk, nullptr, nullptr, khB, MTOT, CC, CC);
    gemm_wmma<1, 0, 0><<<blkCC, 256, 0, stream>>>(vb, Wvb, bv, nullptr, nullptr, vhB, MTOT, CC, CC);

    // Fused flash attention
    attn_wmma<<<BB * HH * (NN / 64), 128, 0, stream>>>(qhB, khB, vhB, ctxB);

    // x = ctx @ Wo + bo + q   (f32 out)
    gemm_wmma<0, 0, 1><<<blkCC, 256, 0, stream>>>(ctxB, Wob, bo, q, xF, nullptr, MTOT, CC, CC);

    // h = LN(x) (bf16 out)
    ln_kernel<<<MTOT, 256, 0, stream>>>(xF, lnw, lnb, hB);

    // g = gelu(h @ W1 + b1) (bf16 out)
    const int blkFF = (MTOT / 128) * (FFD / 256); // 256
    gemm_wmma<1, 1, 0><<<blkFF, 256, 0, stream>>>(hB, W1b, b1, nullptr, nullptr, gB, MTOT, FFD, CC);

    // y = g @ W2 + b2 + x  (f32 out -> d_out)
    gemm_wmma<0, 0, 1><<<blkCC, 256, 0, stream>>>(gB, W2b, b2, xF, out, nullptr, MTOT, CC, FFD);
}
